// LEGCN_50233937494093
// MI455X (gfx1250) — matrix operand order repaired and checked by reading, not compiled, not run on hardware
//
#include <hip/hip_runtime.h>
#include <hip/hip_bf16.h>

typedef __attribute__((ext_vector_type(8)))  float   v8f;
typedef __attribute__((ext_vector_type(16))) __bf16  v16bf;

union V16 {
    v16bf bf;
    uint4 q[2];
};

static __device__ __forceinline__ unsigned short bf16bits(float f) {
    unsigned u = __builtin_bit_cast(unsigned, f);
    unsigned r = u + 0x7FFFu + ((u >> 16) & 1u);   // round-to-nearest-even
    return (unsigned short)(r >> 16);
}

static __device__ __forceinline__ float bf16tof(unsigned short h) {
    return __builtin_bit_cast(float, (unsigned)h << 16);
}

// ---------------------------------------------------------------------------
// Utility kernels
// ---------------------------------------------------------------------------
__global__ void zero_f32(float* __restrict__ p, size_t n) {
    size_t i = (size_t)blockIdx.x * 256 + threadIdx.x;
    if (i < n) p[i] = 0.0f;
}

__global__ void deg_kernel(const int* __restrict__ dst, float* __restrict__ deg, int E) {
    int e = blockIdx.x * 256 + threadIdx.x;
    if (e < E) atomicAdd(&deg[dst[e]], 1.0f);
}

__global__ void cnt_kernel(const int* __restrict__ node_idx, float* __restrict__ cnt, int T) {
    int t = blockIdx.x * 256 + threadIdx.x;
    if (t < T) atomicAdd(&cnt[node_idx[t]], 1.0f);
}

__global__ void dis_kernel(float* __restrict__ deg_dis, int T) {
    int t = blockIdx.x * 256 + threadIdx.x;
    if (t < T) deg_dis[t] = rsqrtf(deg_dis[t] + 1.0f);
}

// ---------------------------------------------------------------------------
// GEMM1: h1[T,128] = bf16(x[node_idx[t],:]) @ bf16(W1[128,128]), f32 accum (WMMA)
// Block: 128 threads (4 waves), 64 FULL rows per block (no bounds checks;
// remainder rows handled by gemm1_tail). Each wave: 16 rows x 128 cols.
// ---------------------------------------------------------------------------
__global__ __launch_bounds__(128) void gemm1_wmma(
        const float* __restrict__ x, const int* __restrict__ node_idx,
        const float* __restrict__ W1, float* __restrict__ h1) {
    __shared__ __align__(16) unsigned short Ws[128][128];  // Ws[n][k] = bf16(W1[k][n])
    __shared__ __align__(16) unsigned short As[64][128];   // gathered bf16 rows

    const int tid = threadIdx.x;
    const int row0 = blockIdx.x * 64;

    // stage W1 transposed (coalesced global reads)
    for (int i = tid; i < 128 * 128; i += 128) {
        int k = i >> 7, n = i & 127;
        Ws[n][k] = bf16bits(W1[i]);
    }
    // stage gathered A rows: thread handles half of one row
    {
        int r = tid >> 1;
        int koff = (tid & 1) * 64;
        int node = node_idx[row0 + r];
        const float* src = x + (size_t)node * 128 + koff;
        #pragma unroll 8
        for (int k = 0; k < 64; ++k) As[r][koff + k] = bf16bits(src[k]);
    }
    __syncthreads();

    const int wave = tid >> 5, lane = tid & 31;
    const int rbase = wave * 16;
    const int lr = lane & 15, lh = lane >> 4;

    for (int nt = 0; nt < 8; ++nt) {
        v8f acc = {};
        const int col = nt * 16 + lr;
        #pragma unroll
        for (int kt = 0; kt < 4; ++kt) {
            const int kb = kt * 32;
            V16 a, b;
            // A 16x32 bf16: lane row=lr; VGPR0-3 hold K = kb + lh*8 + 0..7,
            //               VGPR4-7 hold K = kb + 16 + lh*8 + 0..7
            a.q[0] = *(const uint4*)&As[rbase + lr][kb + lh * 8];
            a.q[1] = *(const uint4*)&As[rbase + lr][kb + 16 + lh * 8];
            // B 32x16 bf16: lane col; VGPR0-3 K = kb + lh*8 + 0..7, VGPR4-7 K = kb + 16 + lh*8 + 0..7
            b.q[0] = *(const uint4*)&Ws[col][kb + lh * 8];
            b.q[1] = *(const uint4*)&Ws[col][kb + 16 + lh * 8];
            acc = __builtin_amdgcn_wmma_f32_16x16x32_bf16(
                      false, a.bf, false, b.bf, (short)0, acc, false, false);
        }
        // D 16x16 f32: VGPR r -> M = 8*lh + r, N = lr  (branch-free stores)
        float* base = h1 + (size_t)(row0 + rbase + 8 * lh) * 128 + nt * 16 + lr;
        #pragma unroll
        for (int r = 0; r < 8; ++r) base[(size_t)r * 128] = acc[r];
    }
}

// scalar tail for rows [t0, T): one block per row, one thread per output col
__global__ __launch_bounds__(128) void gemm1_tail(
        const float* __restrict__ x, const int* __restrict__ node_idx,
        const float* __restrict__ W1, float* __restrict__ h1, int t0) {
    int t = t0 + blockIdx.x;
    int n = threadIdx.x;
    int node = node_idx[t];
    const float* xr = x + (size_t)node * 128;
    float acc = 0.0f;
    for (int k = 0; k < 128; ++k) acc += xr[k] * W1[k * 128 + n];
    h1[(size_t)t * 128 + n] = acc;
}

// ---------------------------------------------------------------------------
// GEMM2: h2[T,64] = h1b(bf16)[T,128] @ bf16(W2[128,64]), f32 accum (WMMA)
// ---------------------------------------------------------------------------
__global__ __launch_bounds__(128) void gemm2_wmma(
        const unsigned short* __restrict__ h1b,
        const float* __restrict__ W2, float* __restrict__ h2) {
    __shared__ __align__(16) unsigned short Ws[64][128];   // Ws[c][k] = bf16(W2[k][c])
    __shared__ __align__(16) unsigned short As[64][128];

    const int tid = threadIdx.x;
    const int row0 = blockIdx.x * 64;

    for (int i = tid; i < 128 * 64; i += 128) {
        int k = i >> 6, c = i & 63;
        Ws[c][k] = bf16bits(W2[i]);
    }
    // A tile already bf16: bulk copy as dwords
    {
        const unsigned int* src = (const unsigned int*)(h1b + (size_t)row0 * 128);
        unsigned int* dst = (unsigned int*)&As[0][0];
        for (int i = tid; i < 64 * 64; i += 128) dst[i] = src[i];
    }
    __syncthreads();

    const int wave = tid >> 5, lane = tid & 31;
    const int rbase = wave * 16;
    const int lr = lane & 15, lh = lane >> 4;

    for (int nt = 0; nt < 4; ++nt) {
        v8f acc = {};
        const int col = nt * 16 + lr;
        #pragma unroll
        for (int kt = 0; kt < 4; ++kt) {
            const int kb = kt * 32;
            V16 a, b;
            a.q[0] = *(const uint4*)&As[rbase + lr][kb + lh * 8];
            a.q[1] = *(const uint4*)&As[rbase + lr][kb + 16 + lh * 8];
            b.q[0] = *(const uint4*)&Ws[col][kb + lh * 8];
            b.q[1] = *(const uint4*)&Ws[col][kb + 16 + lh * 8];
            acc = __builtin_amdgcn_wmma_f32_16x16x32_bf16(
                      false, a.bf, false, b.bf, (short)0, acc, false, false);
        }
        float* base = h2 + (size_t)(row0 + rbase + 8 * lh) * 64 + nt * 16 + lr;
        #pragma unroll
        for (int r = 0; r < 8; ++r) base[(size_t)r * 64] = acc[r];
    }
}

__global__ __launch_bounds__(64) void gemm2_tail(
        const unsigned short* __restrict__ h1b,
        const float* __restrict__ W2, float* __restrict__ h2, int t0) {
    int t = t0 + blockIdx.x;
    int c = threadIdx.x;
    const unsigned short* ar = h1b + (size_t)t * 128;
    float acc = 0.0f;
    for (int k = 0; k < 128; ++k) acc += bf16tof(ar[k]) * W2[k * 64 + c];
    h2[(size_t)t * 64 + c] = acc;
}

// ---------------------------------------------------------------------------
// Edge scatter: agg[dst] += h[src] * (dis[src]*dis[dst]) ; 8 edges/block,
// one wave per edge, float4 (128-wide) or float2 (64-wide) per lane.
// ---------------------------------------------------------------------------
__global__ __launch_bounds__(256) void agg_scatter128(
        const float* __restrict__ h, const float* __restrict__ dis,
        const int* __restrict__ src, const int* __restrict__ dst,
        float* __restrict__ agg, int E) {
    int e = blockIdx.x * 8 + (threadIdx.x >> 5);
    if (e >= E) return;
    int lane = threadIdx.x & 31;
    int s = src[e], d = dst[e];
    // prefetch next edge batch's source row while this one is in flight
    if (e + 8 < E) {
        int sn = src[e + 8];
        __builtin_prefetch(h + (size_t)sn * 128 + lane * 4, 0, 0);
    }
    float nrm = dis[s] * dis[d];
    float4 v = ((const float4*)(h + (size_t)s * 128))[lane];
    float* o = agg + (size_t)d * 128 + lane * 4;
    atomicAdd(o + 0, v.x * nrm);
    atomicAdd(o + 1, v.y * nrm);
    atomicAdd(o + 2, v.z * nrm);
    atomicAdd(o + 3, v.w * nrm);
}

__global__ __launch_bounds__(256) void agg_scatter64(
        const float* __restrict__ h, const float* __restrict__ dis,
        const int* __restrict__ src, const int* __restrict__ dst,
        float* __restrict__ agg, int E) {
    int e = blockIdx.x * 8 + (threadIdx.x >> 5);
    if (e >= E) return;
    int lane = threadIdx.x & 31;
    int s = src[e], d = dst[e];
    if (e + 8 < E) {
        int sn = src[e + 8];
        __builtin_prefetch(h + (size_t)sn * 64 + lane * 2, 0, 0);
    }
    float nrm = dis[s] * dis[d];
    float2 v = ((const float2*)(h + (size_t)s * 64))[lane];
    float* o = agg + (size_t)d * 64 + lane * 2;
    atomicAdd(o + 0, v.x * nrm);
    atomicAdd(o + 1, v.y * nrm);
}

// ---------------------------------------------------------------------------
// post1: h1b = bf16(relu(agg1 + h1*dis^2 + b1))
// ---------------------------------------------------------------------------
__global__ void post1_kernel(const float* __restrict__ h1, const float* __restrict__ agg1,
                             const float* __restrict__ dis, const float* __restrict__ b1,
                             unsigned short* __restrict__ h1b, size_t total) {
    size_t i = (size_t)blockIdx.x * 256 + threadIdx.x;
    if (i >= total) return;
    size_t t = i >> 7;
    int f = (int)(i & 127);
    float dd = dis[t];
    float v = agg1[i] + h1[i] * dd * dd + b1[f];
    v = v > 0.0f ? v : 0.0f;
    h1b[i] = bf16bits(v);
}

// post2: v = agg2 + h2*dis^2 + b2 ; scatter-sum into sum[node_idx[t]]
__global__ void post2_kernel(const float* __restrict__ h2, const float* __restrict__ agg2,
                             const float* __restrict__ dis, const float* __restrict__ b2,
                             const int* __restrict__ node_idx, float* __restrict__ sum,
                             size_t total) {
    size_t i = (size_t)blockIdx.x * 256 + threadIdx.x;
    if (i >= total) return;
    size_t t = i >> 6;
    int c = (int)(i & 63);
    float dd = dis[t];
    float v = agg2[i] + h2[i] * dd * dd + b2[c];
    atomicAdd(&sum[(size_t)node_idx[t] * 64 + c], v);
}

__global__ void finalize_kernel(const float* __restrict__ sum, const float* __restrict__ cnt,
                                float* __restrict__ out, size_t total) {
    size_t i = (size_t)blockIdx.x * 256 + threadIdx.x;
    if (i >= total) return;
    float c = cnt[i >> 6];
    c = c < 1.0f ? 1.0f : c;
    out[i] = sum[i] / c;
}

// ---------------------------------------------------------------------------
extern "C" void kernel_launch(void* const* d_in, const int* in_sizes, int n_in,
                              void* d_out, int out_size, void* d_ws, size_t ws_size,
                              hipStream_t stream) {
    const float* x        = (const float*)d_in[0];
    const float* W1       = (const float*)d_in[1];
    const float* b1       = (const float*)d_in[2];
    const float* W2       = (const float*)d_in[3];
    const float* b2       = (const float*)d_in[4];
    const int*   node_idx = (const int*)d_in[5];
    const int*   le_src   = (const int*)d_in[6];
    const int*   le_dst   = (const int*)d_in[7];
    float* out = (float*)d_out;

    const size_t T = (size_t)in_sizes[5];
    const size_t E = (size_t)in_sizes[6];
    const size_t N = (size_t)in_sizes[0] / 128;

    const size_t full_blocks = T / 64;       // WMMA path covers full 64-row tiles
    const size_t tail_rows   = T - full_blocks * 64;
    const int    t0          = (int)(full_blocks * 64);

    // Workspace layout (bytes). h1 region reused for h2+agg2 after layer 1 dies.
    char* ws = (char*)d_ws;
    float*          h1   = (float*)ws;                                   // T*128 f32
    float*          h2   = (float*)ws;                                   // T*64  f32 (reuse)
    float*          agg2 = (float*)(ws + T * 64 * 4);                    // T*64  f32 (reuse)
    float*          agg1 = (float*)(ws + T * 128 * 4);                   // T*128 f32
    unsigned short* h1b  = (unsigned short*)(ws + T * 128 * 4 * 2);      // T*128 bf16
    float*          dis  = (float*)(ws + T * 128 * 4 * 2 + T * 128 * 2); // T f32 (deg->dis)
    float*          sum  = (float*)((char*)dis + T * 4);                 // N*64 f32
    float*          cnt  = (float*)((char*)sum + N * 64 * 4);            // N f32

    auto blocks = [](size_t n, size_t bs) { return (unsigned)((n + bs - 1) / bs); };

    // 1. zeros
    zero_f32<<<blocks(T, 256), 256, 0, stream>>>(dis, T);
    zero_f32<<<blocks(T * 128, 256), 256, 0, stream>>>(agg1, T * 128);
    zero_f32<<<blocks(N * 64, 256), 256, 0, stream>>>(sum, N * 64);
    zero_f32<<<blocks(N, 256), 256, 0, stream>>>(cnt, N);

    // 2. degrees / counts / normalization
    deg_kernel<<<blocks(E, 256), 256, 0, stream>>>(le_dst, dis, (int)E);
    cnt_kernel<<<blocks(T, 256), 256, 0, stream>>>(node_idx, cnt, (int)T);
    dis_kernel<<<blocks(T, 256), 256, 0, stream>>>(dis, (int)T);

    // 3. layer 1
    if (full_blocks)
        gemm1_wmma<<<(unsigned)full_blocks, 128, 0, stream>>>(x, node_idx, W1, h1);
    if (tail_rows)
        gemm1_tail<<<(unsigned)tail_rows, 128, 0, stream>>>(x, node_idx, W1, h1, t0);
    agg_scatter128<<<blocks(E, 8), 256, 0, stream>>>(h1, dis, le_src, le_dst, agg1, (int)E);
    post1_kernel<<<blocks(T * 128, 256), 256, 0, stream>>>(h1, agg1, dis, b1, h1b, T * 128);

    // 4. layer 2 (h1/agg1 dead from here; region reused for h2/agg2)
    if (full_blocks)
        gemm2_wmma<<<(unsigned)full_blocks, 128, 0, stream>>>(h1b, W2, h2);
    if (tail_rows)
        gemm2_tail<<<(unsigned)tail_rows, 64, 0, stream>>>(h1b, W2, h2, t0);
    zero_f32<<<blocks(T * 64, 256), 256, 0, stream>>>(agg2, T * 64);
    agg_scatter64<<<blocks(E, 8), 256, 0, stream>>>(h2, dis, le_src, le_dst, agg2, (int)E);

    // 5. scatter-mean back to nodes
    post2_kernel<<<blocks(T * 64, 256), 256, 0, stream>>>(h2, agg2, dis, b2, node_idx, sum, T * 64);
    finalize_kernel<<<blocks(N * 64, 256), 256, 0, stream>>>(sum, cnt, out, N * 64);
}
